// GAT_1434519076950
// MI455X (gfx1250) — matrix operand order repaired and checked by reading, not compiled, not run on hardware
//
#include <hip/hip_runtime.h>
#include <math.h>

// ---------------------------------------------------------------------------
// GAT (2-layer) for MI455X / gfx1250.
//   N=50000, IN=256, HEADS=4, HID=64, OUT=64, E=800000 + N self-loops.
//   GEMMs: pre-packed bf16 WMMA fragments -> v_wmma_f32_16x16x32_bf16,
//          fully unrolled K loop (8 WMMA / tile), b128 fragment loads.
//   Edge phase: f32 atomics (max via int/uint trick, add native).
// ---------------------------------------------------------------------------

#define NEG_SLOPE 0.2f

typedef __attribute__((ext_vector_type(16))) __bf16 v16bf;
typedef __attribute__((ext_vector_type(8)))  float  v8f;

__device__ __forceinline__ __bf16 f2bf(float f) {
  union { float f; unsigned u; } v; v.f = f;
  unsigned r = v.u + 0x7FFFu + ((v.u >> 16) & 1u);   // round-to-nearest-even
  unsigned short s = (unsigned short)(r >> 16);
  __bf16 out;
  __builtin_memcpy(&out, &s, sizeof(out));
  return out;
}

// ---------------- fragment packing ----------------------------------------
// A [M,K] f32 row-major -> per-wave A fragments (16-bit A 16x32 layout):
//   frag[((tm*KS + ks)*32 + lane)*16 + e] = A[tm*16 + (lane&15)]
//                                            [ks*32 + 16*(e>>3) + 8*(lane>>4) + (e&7)]
__global__ void pack_A_k(const float* __restrict__ A, __bf16* __restrict__ Ap,
                         int ksteps, int K, int total) {
  int t = blockIdx.x * blockDim.x + threadIdx.x;
  if (t >= total) return;
  int e    = t & 15;
  int lane = (t >> 4) & 31;
  int rest = t >> 9;
  int ks   = rest % ksteps;
  int tm   = rest / ksteps;
  int row  = tm * 16 + (lane & 15);
  int col  = ks * 32 + ((e >> 3) << 4) + ((lane >> 4) << 3) + (e & 7);
  Ap[t] = f2bf(A[(size_t)row * K + col]);
}

// B [K,Nc] f32 row-major -> per-wave B fragments (16-bit B 32x16 layout):
//   frag[((tn*KS + ks)*32 + lane)*16 + e] = B[ks*32 + 16*(e>>3) + 8*(lane>>4) + (e&7)]
//                                            [tn*16 + (lane&15)]
__global__ void pack_B_k(const float* __restrict__ B, __bf16* __restrict__ Bp,
                         int ksteps, int Nc, int total) {
  int t = blockIdx.x * blockDim.x + threadIdx.x;
  if (t >= total) return;
  int e    = t & 15;
  int lane = (t >> 4) & 31;
  int rest = t >> 9;
  int ks   = rest % ksteps;
  int tn   = rest / ksteps;
  int row  = ks * 32 + ((e >> 3) << 4) + ((lane >> 4) << 3) + (e & 7);
  int col  = tn * 16 + (lane & 15);
  Bp[t] = f2bf(B[(size_t)row * Nc + col]);
}

// ---------------- WMMA GEMM on packed fragments ----------------------------
// C[M,Nc] = A*B, one wave per 16x16 C tile, K compile-time -> unrolled WMMAs.
template <int K>
__global__ void gemm_wmma_packed(const __bf16* __restrict__ Ap,
                                 const __bf16* __restrict__ Bp,
                                 float* __restrict__ C,
                                 int Nc, int total_tiles) {
  constexpr int KS = K / 32;
  const int wave = (int)((blockIdx.x * blockDim.x + threadIdx.x) >> 5);
  if (wave >= total_tiles) return;                 // wave-uniform: EXEC all-1s
  const int lane   = threadIdx.x & 31;
  const int ntiles = Nc >> 4;
  const int tm = wave / ntiles;
  const int tn = wave - tm * ntiles;

  const v16bf* Af = (const v16bf*)Ap + ((size_t)tm * KS * 32 + lane);
  const v16bf* Bf = (const v16bf*)Bp + ((size_t)tn * KS * 32 + lane);

  v8f acc = {};
#pragma unroll
  for (int ks = 0; ks < KS; ++ks) {
    v16bf a = Af[(size_t)ks * 32];                 // 32B -> 2x global_load_b128
    v16bf b = Bf[(size_t)ks * 32];
    acc = __builtin_amdgcn_wmma_f32_16x16x32_bf16(false, a, false, b,
                                                  (short)0, acc, false, false);
  }
  // C/D layout: VGPR r -> row 8*(lane>>4) + r, col = lane&15
  const int half = lane >> 4, l16 = lane & 15;
  float* Cout = C + (size_t)(tm * 16 + (half << 3)) * Nc + tn * 16 + l16;
#pragma unroll
  for (int r = 0; r < 8; ++r) Cout[(size_t)r * Nc] = acc[r];
}

// ---------------- utility fills -------------------------------------------
__global__ void fill_f32(float* __restrict__ p, float v, int n) {
  int i = blockIdx.x * blockDim.x + threadIdx.x;
  if (i < n) p[i] = v;
}

// ---------------- attention scores: a_s[n,h] = <h[n,h,:], att_s[h,:]> ------
template <int H, int C>
__global__ void attn_scores_k(const float* __restrict__ h,
                              const float* __restrict__ att_s,
                              const float* __restrict__ att_d,
                              float* __restrict__ a_s,
                              float* __restrict__ a_d, int n) {
  int t = blockIdx.x * blockDim.x + threadIdx.x;      // t = node*H + head
  if (t >= n * H) return;
  int hh = t % H;
  const float* row = h + (size_t)t * C;               // h is [n, H*C] row-major
  float ss = 0.f, sd = 0.f;
#pragma unroll 8
  for (int c = 0; c < C; ++c) {
    float x = row[c];
    ss += x * att_s[hh * C + c];
    sd += x * att_d[hh * C + c];
  }
  a_s[t] = ss;
  a_d[t] = sd;
}

// ---------------- float atomic max via int/uint ordering trick -------------
__device__ __forceinline__ void atomic_max_f32(float* addr, float val) {
  if (val >= 0.f)
    atomicMax((int*)addr, __float_as_int(val));
  else
    atomicMin((unsigned int*)addr, __float_as_uint(val));
}

__device__ __forceinline__ float leaky(float v) {
  return v > 0.f ? v : NEG_SLOPE * v;
}

// ---------------- edge pass 1: segment max of leaky_relu scores ------------
template <int H>
__global__ void edge_max_k(const int* __restrict__ src, const int* __restrict__ dst,
                           int E0, int Etot,
                           const float* __restrict__ a_s, const float* __restrict__ a_d,
                           float* __restrict__ m) {
  int t = blockIdx.x * blockDim.x + threadIdx.x;
  if (t >= Etot * H) return;
  int e = t / H, hh = t - e * H;
  int s, d;
  if (e < E0) { s = src[e]; d = dst[e]; } else { s = e - E0; d = s; }
  float v = leaky(a_s[s * H + hh] + a_d[d * H + hh]);
  atomic_max_f32(&m[d * H + hh], v);
}

// ---------------- edge pass 2: denom = segment sum of exp(e - m[dst]) ------
template <int H>
__global__ void edge_expsum_k(const int* __restrict__ src, const int* __restrict__ dst,
                              int E0, int Etot,
                              const float* __restrict__ a_s, const float* __restrict__ a_d,
                              const float* __restrict__ m, float* __restrict__ den) {
  int t = blockIdx.x * blockDim.x + threadIdx.x;
  if (t >= Etot * H) return;
  int e = t / H, hh = t - e * H;
  int s, d;
  if (e < E0) { s = src[e]; d = dst[e]; } else { s = e - E0; d = s; }
  float v = leaky(a_s[s * H + hh] + a_d[d * H + hh]);
  atomicAdd(&den[d * H + hh], __expf(v - m[d * H + hh]));
}

// ---------------- edge pass 3: out[dst] += alpha * h[src] ------------------
// One wave32 per edge; lanes stride the H*C feature row (coalesced gather +
// coalesced atomic scatter).
template <int H, int C>
__global__ void edge_aggregate_k(const int* __restrict__ src, const int* __restrict__ dst,
                                 int E0, int Etot,
                                 const float* __restrict__ h,
                                 const float* __restrict__ a_s, const float* __restrict__ a_d,
                                 const float* __restrict__ m, const float* __restrict__ den,
                                 float* __restrict__ out) {
  const int w = (int)((blockIdx.x * blockDim.x + threadIdx.x) >> 5);
  if (w >= Etot) return;
  const int lane = threadIdx.x & 31;
  int s, d;
  if (w < E0) { s = src[w]; d = dst[w]; } else { s = w - E0; d = s; }

  float alpha[H];
#pragma unroll
  for (int hh = 0; hh < H; ++hh) {
    float v = leaky(a_s[s * H + hh] + a_d[d * H + hh]);
    float ex = __expf(v - m[d * H + hh]);
    alpha[hh] = ex / (den[d * H + hh] + 1e-16f);
  }
  constexpr int HC = H * C;
#pragma unroll
  for (int i = 0; i < HC / 32; ++i) {
    int idx = lane + 32 * i;
    int hh = idx / C;
    atomicAdd(&out[(size_t)d * HC + idx], alpha[hh] * h[(size_t)s * HC + idx]);
  }
}

// ---------------- epilogues ------------------------------------------------
__global__ void bias_elu_k(float* __restrict__ x, const float* __restrict__ b,
                           int n, int C) {
  int i = blockIdx.x * blockDim.x + threadIdx.x;
  if (i >= n) return;
  float v = x[i] + b[i % C];
  x[i] = v > 0.f ? v : expm1f(v);
}

__global__ void bias_add_k(float* __restrict__ x, const float* __restrict__ b,
                           int n, int C) {
  int i = blockIdx.x * blockDim.x + threadIdx.x;
  if (i >= n) return;
  x[i] += b[i % C];
}

// ---------------------------------------------------------------------------
extern "C" void kernel_launch(void* const* d_in, const int* in_sizes, int n_in,
                              void* d_out, int out_size, void* d_ws, size_t ws_size,
                              hipStream_t stream) {
  const float* x    = (const float*)d_in[0];
  const int*   ei   = (const int*)d_in[1];   // [2, E0] int32 (jax default x64 off)
  const float* W1   = (const float*)d_in[2];
  const float* av_s1= (const float*)d_in[3];
  const float* av_d1= (const float*)d_in[4];
  const float* b1   = (const float*)d_in[5];
  const float* W2   = (const float*)d_in[6];
  const float* av_s2= (const float*)d_in[7];
  const float* av_d2= (const float*)d_in[8];
  const float* b2   = (const float*)d_in[9];
  float* out = (float*)d_out;

  const int N  = in_sizes[0] / 256;     // 50000 (divisible by 16)
  const int E0 = in_sizes[1] / 2;       // 800000
  const int E  = E0 + N;                // + self loops
  const int* srcp = ei;
  const int* dstp = ei + E0;

  // workspace carve-up, 256-byte aligned blocks (~145 MB total)
  char* wsb = (char*)d_ws;
  auto carve = [&wsb](size_t bytes) -> char* {
    char* p = wsb;
    wsb += (bytes + 255) & ~(size_t)255;
    return p;
  };
  float*  h1    = (float*)carve((size_t)N * 256 * 4);  // layer-1 features
  float*  x2    = (float*)carve((size_t)N * 256 * 4);  // aggregate -> elu -> L2 input
  float*  as1   = (float*)carve((size_t)N * 4 * 4);
  float*  ad1   = (float*)carve((size_t)N * 4 * 4);
  float*  m1    = (float*)carve((size_t)N * 4 * 4);
  float*  dn1   = (float*)carve((size_t)N * 4 * 4);
  float*  h2    = (float*)carve((size_t)N * 64 * 4);   // layer-2 features
  float*  as2   = (float*)carve((size_t)N * 4);
  float*  ad2   = (float*)carve((size_t)N * 4);
  float*  m2    = (float*)carve((size_t)N * 4);
  float*  dn2   = (float*)carve((size_t)N * 4);
  __bf16* Apack = (__bf16*)carve((size_t)N * 256 * 2); // reused by both layers
  __bf16* Bpack = (__bf16*)carve((size_t)256 * 256 * 2);

  const int TB = 256;                        // 8 waves / block (wave32)
  auto nb = [](long n, int tb) { return (int)((n + tb - 1) / tb); };
  const float NEG_INF = -3.402823466e38f;

  // ---- init (harness does not re-poison between replays) ----
  fill_f32<<<nb((long)N * 256, TB), TB, 0, stream>>>(x2, 0.f, N * 256);
  fill_f32<<<nb((long)N * 4, TB), TB, 0, stream>>>(m1, NEG_INF, N * 4);
  fill_f32<<<nb((long)N * 4, TB), TB, 0, stream>>>(dn1, 0.f, N * 4);
  fill_f32<<<nb(N, TB), TB, 0, stream>>>(m2, NEG_INF, N);
  fill_f32<<<nb(N, TB), TB, 0, stream>>>(dn2, 0.f, N);
  fill_f32<<<nb((long)N * 64, TB), TB, 0, stream>>>(out, 0.f, N * 64);

  // ---- layer 1: h1 = x @ W1 via packed WMMA ----
  pack_A_k<<<nb((long)N * 256, TB), TB, 0, stream>>>(x, Apack, 8, 256, N * 256);
  pack_B_k<<<nb(256 * 256, TB), TB, 0, stream>>>(W1, Bpack, 8, 256, 256 * 256);
  {
    int tiles = (N / 16) * (256 / 16);       // 50000 tiles, 1 wave each
    gemm_wmma_packed<256><<<nb((long)tiles * 32, TB), TB, 0, stream>>>(
        Apack, Bpack, h1, 256, tiles);
  }
  attn_scores_k<4, 64><<<nb((long)N * 4, TB), TB, 0, stream>>>(h1, av_s1, av_d1, as1, ad1, N);
  edge_max_k<4><<<nb((long)E * 4, TB), TB, 0, stream>>>(srcp, dstp, E0, E, as1, ad1, m1);
  edge_expsum_k<4><<<nb((long)E * 4, TB), TB, 0, stream>>>(srcp, dstp, E0, E, as1, ad1, m1, dn1);
  edge_aggregate_k<4, 64><<<nb((long)E * 32, TB), TB, 0, stream>>>(
      srcp, dstp, E0, E, h1, as1, ad1, m1, dn1, x2);
  bias_elu_k<<<nb((long)N * 256, TB), TB, 0, stream>>>(x2, b1, N * 256, 256);

  // ---- layer 2: h2 = x2 @ W2 via packed WMMA ----
  pack_A_k<<<nb((long)N * 256, TB), TB, 0, stream>>>(x2, Apack, 8, 256, N * 256);
  pack_B_k<<<nb(256 * 64, TB), TB, 0, stream>>>(W2, Bpack, 8, 64, 256 * 64);
  {
    int tiles = (N / 16) * (64 / 16);        // 12500 tiles
    gemm_wmma_packed<256><<<nb((long)tiles * 32, TB), TB, 0, stream>>>(
        Apack, Bpack, h2, 64, tiles);
  }
  attn_scores_k<1, 64><<<nb(N, TB), TB, 0, stream>>>(h2, av_s2, av_d2, as2, ad2, N);
  edge_max_k<1><<<nb(E, TB), TB, 0, stream>>>(srcp, dstp, E0, E, as2, ad2, m2);
  edge_expsum_k<1><<<nb(E, TB), TB, 0, stream>>>(srcp, dstp, E0, E, as2, ad2, m2, dn2);
  edge_aggregate_k<1, 64><<<nb((long)E * 32, TB), TB, 0, stream>>>(
      srcp, dstp, E0, E, h2, as2, ad2, m2, dn2, out);
  bias_add_k<<<nb((long)N * 64, TB), TB, 0, stream>>>(out, b2, N * 64, 64);
}
